// EncoderLayer_13778255086263
// MI455X (gfx1250) — matrix operand order repaired and checked by reading, not compiled, not run on hardware
//
#include <hip/hip_runtime.h>
#include <hip/hip_bf16.h>

typedef __attribute__((ext_vector_type(16))) _Float16 v16h;
typedef __attribute__((ext_vector_type(8)))  _Float16 v8h;
typedef __attribute__((ext_vector_type(8)))  float    v8f;

#define BATCH   4
#define SEQ     2048
#define DM      256
#define NHEAD   8
#define DEPTH   32
#define DFF     1024
#define NROWS   (BATCH * SEQ)     // 8192

// ---- CDNA5 async-to-LDS helpers (inline asm; portable across toolchains) ----
// global_load_async_to_lds_b128: VDST = VGPR holding LDS byte address,
// VADDR = 32-bit byte offset VGPR, SADDR = 64-bit base SGPR pair. ASYNCcnt.
__device__ __forceinline__ void async_ld_b128(unsigned lds_addr, unsigned gl_off,
                                              unsigned long long base) {
  asm volatile("global_load_async_to_lds_b128 %0, %1, %2"
               :: "v"(lds_addr), "v"(gl_off), "s"(base) : "memory");
}
__device__ __forceinline__ void wait_async0() {
  asm volatile("s_wait_asynccnt 0" ::: "memory");
}
// low 32 bits of a generic pointer to LDS == byte offset into the workgroup's
// LDS allocation
__device__ __forceinline__ unsigned lds_off(const void* p) {
  return (unsigned)(size_t)p;
}

// ---------------- small helper kernels ----------------

__global__ __launch_bounds__(256) void k_to_f16(const float* __restrict__ src,
                                                _Float16* __restrict__ dst, int n) {
  int i = blockIdx.x * 256 + threadIdx.x;
  if (i < n) dst[i] = (_Float16)src[i];
}

// dst[N,K] = (f16) src[K,N]^T   (weights -> NT-GEMM B operand)
__global__ __launch_bounds__(256) void k_transpose_f16(const float* __restrict__ src,
                                                       _Float16* __restrict__ dst,
                                                       int K, int N) {
  int i = blockIdx.x * 256 + threadIdx.x;
  if (i >= K * N) return;
  int n = i / K, k = i % K;
  dst[i] = (_Float16)src[(size_t)k * N + n];
}

// ---------------- generic NT GEMM: C = A[M,K] * Bt[N,K]^T + bias ----------------
// modes: 0 = f32 out ; 1 = f16 out ; 2 = f16 split-head [B,H,S,32] ; 3 = f16 Vt [B,H,32,S]
// Staging uses GLOBAL_LOAD_ASYNC_TO_LDS_B128 with double-buffered LDS.

#define BM 64
#define BN 128
#define BK 32
#define LDS_STRIDE (BK + 8)   // 40 f16 = 80B rows, keeps 16B alignment

__global__ __launch_bounds__(256) void k_gemm_nt(
    const _Float16* __restrict__ A, const _Float16* __restrict__ Bt,
    const float* __restrict__ bias,
    float* __restrict__ Cf32, _Float16* __restrict__ Cf16,
    int M, int N, int K, int mode, int relu) {
  __shared__ _Float16 As[2][BM * LDS_STRIDE];
  __shared__ _Float16 Bs[2][BN * LDS_STRIDE];

  const int tid  = threadIdx.x;
  const int lane = tid & 31;
  const int wv   = tid >> 5;        // 0..7
  const int hf   = lane >> 4;       // lane half (wave32)
  const int l16  = lane & 15;
  const int tileM = blockIdx.y * BM;
  const int tileN = blockIdx.x * BN;
  const int rtile = wv & 3;         // which 16-row strip
  const int cgrp  = (wv >> 2) * 4;  // 4 consecutive 16-col tiles

  const int arow = tid >> 2, acol = (tid & 3) * 8;   // A tile: 64x32, 8 f16/thread
  const int brow = tid >> 1, bcol = (tid & 1) * 16;  // B tile: 128x32, 16 f16/thread

  const unsigned long long Abase = (unsigned long long)A;
  const unsigned long long Bbase = (unsigned long long)Bt;
  const unsigned aGl0 = (unsigned)(((size_t)(tileM + arow) * K + acol) * 2);
  const unsigned bGl0 = (unsigned)(((size_t)(tileN + brow) * K + bcol) * 2);
  unsigned aLds[2], bLds[2];
  aLds[0] = lds_off(&As[0][arow * LDS_STRIDE + acol]);
  aLds[1] = lds_off(&As[1][arow * LDS_STRIDE + acol]);
  bLds[0] = lds_off(&Bs[0][brow * LDS_STRIDE + bcol]);
  bLds[1] = lds_off(&Bs[1][brow * LDS_STRIDE + bcol]);

  v8f acc[4] = {};

  // prologue: stage tile 0 into buffer 0 (async, bypasses VGPRs)
  async_ld_b128(aLds[0], aGl0, Abase);
  async_ld_b128(bLds[0], bGl0, Bbase);
  async_ld_b128(bLds[0] + 32, bGl0 + 32, Bbase);
  wait_async0();
  __syncthreads();

  int buf = 0;
  for (int k0 = 0; k0 < K; k0 += BK, buf ^= 1) {
    // issue async staging of the NEXT k-tile into the other buffer
    if (k0 + BK < K) {
      const unsigned kadv = (unsigned)(k0 + BK) * 2;
      async_ld_b128(aLds[buf ^ 1], aGl0 + kadv, Abase);
      async_ld_b128(bLds[buf ^ 1], bGl0 + kadv, Bbase);
      async_ld_b128(bLds[buf ^ 1] + 32, bGl0 + kadv + 32, Bbase);
    }

    // compute on the current buffer
    const _Float16* as = &As[buf][0];
    const _Float16* bs = &Bs[buf][0];
    v16h af;
    {
      const _Float16* p = as + (rtile * 16 + l16) * LDS_STRIDE;
      v8h lo = *(const v8h*)(p + hf * 8);
      v8h hi = *(const v8h*)(p + 16 + hf * 8);
      for (int i = 0; i < 8; ++i) { af[i] = lo[i]; af[8 + i] = hi[i]; }
    }
    for (int ct = 0; ct < 4; ++ct) {
      const _Float16* p = bs + ((cgrp + ct) * 16 + l16) * LDS_STRIDE + hf * 8;
      v8h lo = *(const v8h*)p;
      v8h hi = *(const v8h*)(p + 16);
      v16h bf;
      for (int i = 0; i < 8; ++i) { bf[i] = lo[i]; bf[8 + i] = hi[i]; }
      acc[ct] = __builtin_amdgcn_wmma_f32_16x16x32_f16(false, af, false, bf,
                                                       (short)0, acc[ct], false, false);
    }

    // next tile's async writes must land before anyone reads that buffer
    wait_async0();
    __syncthreads();
  }

  for (int ct = 0; ct < 4; ++ct) {
    const int col = tileN + (cgrp + ct) * 16 + l16;
    const float bv = bias ? bias[col] : 0.f;
    for (int r = 0; r < 8; ++r) {
      const int row = tileM + rtile * 16 + hf * 8 + r;
      float v = acc[ct][r] + bv;
      if (relu) v = v > 0.f ? v : 0.f;
      if (mode == 0) {
        Cf32[(size_t)row * N + col] = v;
      } else if (mode == 1) {
        Cf16[(size_t)row * N + col] = (_Float16)v;
      } else {
        const int b = row >> 11, s = row & (SEQ - 1);
        const int h = col >> 5,  d = col & (DEPTH - 1);
        if (mode == 2)
          Cf16[(((size_t)(b * NHEAD + h)) * SEQ + s) * DEPTH + d] = (_Float16)v;
        else
          Cf16[(((size_t)(b * NHEAD + h)) * DEPTH + d) * SEQ + s] = (_Float16)v;
      }
    }
  }
}

// ---------------- fused streaming attention ----------------
// attn = softmax_k( exp(-0.5*(||q||^2 - 2 q.k + ||k||^2)) ),  out = attn @ V
// one wave per 16 queries; keys streamed in blocks of 32; 4 WMMA per step.

__global__ __launch_bounds__(256) void k_attention(
    const _Float16* __restrict__ q16,  // [B,H,S,32]
    const _Float16* __restrict__ k16,  // [B,H,S,32]
    const _Float16* __restrict__ vT,   // [B,H,32,S]
    _Float16* __restrict__ att16) {    // [B,S,256]  (heads merged)
  __shared__ _Float16 P[8][16 * 40];   // per-wave 16x32 weight tile, padded rows

  const int lane = threadIdx.x & 31;
  const int wv   = threadIdx.x >> 5;
  const int hf   = lane >> 4;
  const int l16  = lane & 15;
  const int wid  = blockIdx.x * 8 + wv;
  const int qb   = wid & (SEQ / 16 - 1);
  const int bh   = wid >> 7;                       // 0 .. B*H-1

  const _Float16* qp = q16 + ((size_t)bh * SEQ + qb * 16) * DEPTH;
  const _Float16* kp = k16 + (size_t)bh * SEQ * DEPTH;
  const _Float16* vp = vT  + (size_t)bh * DEPTH * SEQ;

  // Q fragment (A operand, stays resident all loop)
  v16h qa;
  {
    const _Float16* p = qp + l16 * DEPTH + hf * 8;
    v8h lo = *(const v8h*)p;
    v8h hi = *(const v8h*)(p + 16);
    for (int i = 0; i < 8; ++i) { qa[i] = lo[i]; qa[8 + i] = hi[i]; }
  }
  // ||q||^2 per row from the fragment halves
  float q2p = 0.f;
  for (int i = 0; i < 16; ++i) { float t = (float)qa[i]; q2p += t * t; }
  const float q2full = q2p + __shfl_xor(q2p, 16, 32);
  float q2r[8];
  for (int r = 0; r < 8; ++r) q2r[r] = __shfl(q2full, hf * 8 + r, 32);

  v8f o0 = {}, o1 = {};
  float rs[8] = {};
  _Float16* Pw = &P[wv][0];

  for (int kb = 0; kb < SEQ; kb += 32) {
    // K fragments (B operand, rows = keys, cols = depth -> already NT layout)
    v16h bk0, bk1;
    {
      const _Float16* p0 = kp + (size_t)(kb + l16) * DEPTH + hf * 8;
      v8h lo = *(const v8h*)p0, hi = *(const v8h*)(p0 + 16);
      for (int i = 0; i < 8; ++i) { bk0[i] = lo[i]; bk0[8 + i] = hi[i]; }
      const _Float16* p1 = p0 + 16 * DEPTH;
      lo = *(const v8h*)p1; hi = *(const v8h*)(p1 + 16);
      for (int i = 0; i < 8; ++i) { bk1[i] = lo[i]; bk1[8 + i] = hi[i]; }
    }
    // ||k||^2 from fragments (two halves of the same key combine via xor-16)
    float k2p0 = 0.f, k2p1 = 0.f;
    for (int i = 0; i < 16; ++i) {
      float a = (float)bk0[i]; k2p0 += a * a;
      float b = (float)bk1[i]; k2p1 += b * b;
    }
    const float k2a = k2p0 + __shfl_xor(k2p0, 16, 32);
    const float k2b = k2p1 + __shfl_xor(k2p1, 16, 32);

    v8f zc = {};
    v8f s0 = __builtin_amdgcn_wmma_f32_16x16x32_f16(false, qa, false, bk0, (short)0, zc, false, false);
    v8f s1 = __builtin_amdgcn_wmma_f32_16x16x32_f16(false, qa, false, bk1, (short)0, zc, false, false);

    // weights w = exp(exp(-0.5*dist)); inner value in (0,1] so no max trick needed
    for (int r = 0; r < 8; ++r) {
      float d0 = q2r[r] + k2a - 2.f * s0[r];
      float d1 = q2r[r] + k2b - 2.f * s1[r];
      float w0 = __expf(__expf(-0.5f * d0));
      float w1 = __expf(__expf(-0.5f * d1));
      rs[r] += w0 + w1;
      Pw[(hf * 8 + r) * 40 + l16]      = (_Float16)w0;  // C-layout -> LDS
      Pw[(hf * 8 + r) * 40 + 16 + l16] = (_Float16)w1;
    }
    // re-read as A fragment (per-wave LDS region; DS ops in-order within wave)
    v16h pa;
    {
      const _Float16* p = Pw + l16 * 40;
      v8h lo = *(const v8h*)(p + hf * 8);
      v8h hi = *(const v8h*)(p + 16 + hf * 8);
      for (int i = 0; i < 8; ++i) { pa[i] = lo[i]; pa[8 + i] = hi[i]; }
    }
    // V fragments (Vt rows = depth, cols = keys)
    v16h bv0, bv1;
    {
      const _Float16* p0 = vp + (size_t)l16 * SEQ + kb + hf * 8;
      v8h lo = *(const v8h*)p0, hi = *(const v8h*)(p0 + 16);
      for (int i = 0; i < 8; ++i) { bv0[i] = lo[i]; bv0[8 + i] = hi[i]; }
      const _Float16* p1 = vp + (size_t)(16 + l16) * SEQ + kb + hf * 8;
      lo = *(const v8h*)p1; hi = *(const v8h*)(p1 + 16);
      for (int i = 0; i < 8; ++i) { bv1[i] = lo[i]; bv1[8 + i] = hi[i]; }
    }
    o0 = __builtin_amdgcn_wmma_f32_16x16x32_f16(false, pa, false, bv0, (short)0, o0, false, false);
    o1 = __builtin_amdgcn_wmma_f32_16x16x32_f16(false, pa, false, bv1, (short)0, o1, false, false);
  }

  // reduce row sums across the 16 lanes of each half
  for (int r = 0; r < 8; ++r) {
    float v = rs[r];
    v += __shfl_xor(v, 1, 32);
    v += __shfl_xor(v, 2, 32);
    v += __shfl_xor(v, 4, 32);
    v += __shfl_xor(v, 8, 32);
    rs[r] = v;
  }

  const int b = bh >> 3, h = bh & 7;
  for (int r = 0; r < 8; ++r) {
    const int row = qb * 16 + hf * 8 + r;
    const float inv = 1.f / rs[r];
    const size_t base = ((size_t)b * SEQ + row) * DM + h * DEPTH;
    att16[base + l16]      = (_Float16)(o0[r] * inv);
    att16[base + 16 + l16] = (_Float16)(o1[r] * inv);
  }
}

// ---------------- residual + LayerNorm (one wave per 256-wide row) ----------------

__global__ __launch_bounds__(256) void k_ln_residual(
    const float* __restrict__ xa, const float* __restrict__ xb,
    const float* __restrict__ g, const float* __restrict__ bb,
    float* __restrict__ of32, _Float16* __restrict__ of16, int nrows) {
  const int lane = threadIdx.x & 31;
  const int wv   = threadIdx.x >> 5;
  const int row  = blockIdx.x * 8 + wv;
  if (row >= nrows) return;
  const float* pa = xa + (size_t)row * DM;
  const float* pb = xb + (size_t)row * DM;
  float vals[8];
  float s = 0.f;
  for (int j = 0; j < 8; ++j) {
    float v = pa[lane + 32 * j] + pb[lane + 32 * j];
    vals[j] = v; s += v;
  }
  for (int m = 1; m < 32; m <<= 1) s += __shfl_xor(s, m, 32);
  const float mu = s * (1.f / DM);
  float vs = 0.f;
  for (int j = 0; j < 8; ++j) { float d = vals[j] - mu; vs += d * d; }
  for (int m = 1; m < 32; m <<= 1) vs += __shfl_xor(vs, m, 32);
  const float rstd = rsqrtf(vs * (1.f / DM) + 1e-6f);
  for (int j = 0; j < 8; ++j) {
    const int c = lane + 32 * j;
    float o = (vals[j] - mu) * rstd * g[c] + bb[c];
    if (of32) of32[(size_t)row * DM + c] = o;
    if (of16) of16[(size_t)row * DM + c] = (_Float16)o;
  }
}

// ---------------- host-side orchestration ----------------

extern "C" void kernel_launch(void* const* d_in, const int* in_sizes, int n_in,
                              void* d_out, int out_size, void* d_ws, size_t ws_size,
                              hipStream_t stream) {
  const float* x      = (const float*)d_in[0];
  const float* wq_w   = (const float*)d_in[1];  const float* wq_b  = (const float*)d_in[2];
  const float* wk_w   = (const float*)d_in[3];  const float* wk_b  = (const float*)d_in[4];
  const float* wv_w   = (const float*)d_in[5];  const float* wv_b  = (const float*)d_in[6];
  const float* wo_w   = (const float*)d_in[7];  const float* wo_b  = (const float*)d_in[8];
  const float* ffn1_w = (const float*)d_in[9];  const float* ffn1_b= (const float*)d_in[10];
  const float* ffn2_w = (const float*)d_in[11]; const float* ffn2_b= (const float*)d_in[12];
  const float* ln1_g  = (const float*)d_in[13]; const float* ln1_b = (const float*)d_in[14];
  const float* ln2_g  = (const float*)d_in[15]; const float* ln2_b = (const float*)d_in[16];

  char* ws = (char*)d_ws;
  auto alloc = [&](size_t bytes) -> char* {
    char* p = ws;
    ws += (bytes + 255) & ~(size_t)255;
    return p;
  };
  _Float16* x16   = (_Float16*)alloc((size_t)NROWS * DM * 2);
  _Float16* q16   = (_Float16*)alloc((size_t)NROWS * DM * 2);
  _Float16* k16   = (_Float16*)alloc((size_t)NROWS * DM * 2);
  _Float16* vT16  = (_Float16*)alloc((size_t)NROWS * DM * 2);
  _Float16* att16 = (_Float16*)alloc((size_t)NROWS * DM * 2);
  _Float16* out1h = (_Float16*)alloc((size_t)NROWS * DM * 2);
  _Float16* h16   = (_Float16*)alloc((size_t)NROWS * DFF * 2);
  _Float16* wqT   = (_Float16*)alloc((size_t)DM * DM * 2);
  _Float16* wkT   = (_Float16*)alloc((size_t)DM * DM * 2);
  _Float16* wvT   = (_Float16*)alloc((size_t)DM * DM * 2);
  _Float16* woT   = (_Float16*)alloc((size_t)DM * DM * 2);
  _Float16* f1T   = (_Float16*)alloc((size_t)DM * DFF * 2);
  _Float16* f2T   = (_Float16*)alloc((size_t)DFF * DM * 2);
  float*    attnF = (float*)alloc((size_t)NROWS * DM * 4);
  float*    out1F = (float*)alloc((size_t)NROWS * DM * 4);
  float*    ffnF  = (float*)alloc((size_t)NROWS * DM * 4);
  (void)ws_size; (void)in_sizes; (void)n_in; (void)out_size;

  // weight + activation precision conversion
  const int nW = DM * DM, nF = DM * DFF;
  k_transpose_f16<<<(nW + 255) / 256, 256, 0, stream>>>(wq_w, wqT, DM, DM);
  k_transpose_f16<<<(nW + 255) / 256, 256, 0, stream>>>(wk_w, wkT, DM, DM);
  k_transpose_f16<<<(nW + 255) / 256, 256, 0, stream>>>(wv_w, wvT, DM, DM);
  k_transpose_f16<<<(nW + 255) / 256, 256, 0, stream>>>(wo_w, woT, DM, DM);
  k_transpose_f16<<<(nF + 255) / 256, 256, 0, stream>>>(ffn1_w, f1T, DM, DFF);
  k_transpose_f16<<<(nF + 255) / 256, 256, 0, stream>>>(ffn2_w, f2T, DFF, DM);
  k_to_f16<<<(NROWS * DM + 255) / 256, 256, 0, stream>>>(x, x16, NROWS * DM);

  // QKV projections (split-head / transposed-V epilogues)
  dim3 gP(DM / BN, NROWS / BM);
  k_gemm_nt<<<gP, 256, 0, stream>>>(x16, wqT, wq_b, nullptr, q16,  NROWS, DM, DM, 2, 0);
  k_gemm_nt<<<gP, 256, 0, stream>>>(x16, wkT, wk_b, nullptr, k16,  NROWS, DM, DM, 2, 0);
  k_gemm_nt<<<gP, 256, 0, stream>>>(x16, wvT, wv_b, nullptr, vT16, NROWS, DM, DM, 3, 0);

  // fused streaming attention (B*H*S/16 waves, 8 waves per block)
  k_attention<<<BATCH * NHEAD * (SEQ / 16) / 8, 256, 0, stream>>>(q16, k16, vT16, att16);

  // output projection + residual/LN1
  k_gemm_nt<<<gP, 256, 0, stream>>>(att16, woT, wo_b, attnF, nullptr, NROWS, DM, DM, 0, 0);
  k_ln_residual<<<NROWS / 8, 256, 0, stream>>>(x, attnF, ln1_g, ln1_b, out1F, out1h, NROWS);

  // FFN
  dim3 gF1(DFF / BN, NROWS / BM);
  k_gemm_nt<<<gF1, 256, 0, stream>>>(out1h, f1T, ffn1_b, nullptr, h16, NROWS, DFF, DM, 1, 1);
  k_gemm_nt<<<gP, 256, 0, stream>>>(h16, f2T, ffn2_b, ffnF, nullptr, NROWS, DM, DFF, 0, 0);

  // residual/LN2 -> final output
  k_ln_residual<<<NROWS / 8, 256, 0, stream>>>(out1F, ffnF, ln2_g, ln2_b, (float*)d_out, nullptr, NROWS);
}
